// MambaVsimple_36043365548483
// MI455X (gfx1250) — compile-verified, matrix operand order checked by reading, not run on hardware
//
#include <hip/hip_runtime.h>
#include <hip/hip_bf16.h>
#include <cstdint>
#include <cstddef>

typedef __bf16 v16bf __attribute__((ext_vector_type(16)));
typedef __bf16 v8bf  __attribute__((ext_vector_type(8)));
typedef float  v8f   __attribute__((ext_vector_type(8)));
typedef uint32_t u32x4 __attribute__((ext_vector_type(4)));
typedef uint32_t u32x8 __attribute__((ext_vector_type(8)));

#define D_MODEL 1024
#define D_STATE 16
#define D_INNER 2048
#define DT_RANK 64
#define B_SZ    2
#define L_SEQ   1024
#define M_ROWS  (B_SZ * L_SEQ)   // 2048 rows in every GEMM

// LDS tile row stride in bf16 elements: 144 bytes.
//  - multiple of 16B (required for async b128 writes / TDM pad granularity)
//  - 36*r mod 64 banks distinct over the 16 rows of a fragment read
#define LSTR 72

// ---------------------------------------------------------------------------
// fp32 -> bf16 pack, with zero padding in [count, padCount).
// ---------------------------------------------------------------------------
__global__ __launch_bounds__(256)
void pack_bf16(const float* __restrict__ src, __bf16* __restrict__ dst,
               int count, int padCount)
{
    const int idx = blockIdx.x * 256 + threadIdx.x;
    if (idx < padCount)
        dst[idx] = (idx < count) ? (__bf16)src[idx] : (__bf16)0.0f;
}

// ---------------------------------------------------------------------------
// TDM: issue one 2-D tensor_load_to_lds moving a 128-row x 64B tile into LDS.
// data_size = 4B; tile_dim0 = 16 DWORDs (64B), tile_dim1 = 128 rows.
// pad_enable with interval 16 DWORDs (64B) and amount 20 DWORDs (80B)
// reproduces the 144B padded LDS row pitch the WMMA fragment reads use.
// ---------------------------------------------------------------------------
__device__ __forceinline__ void tdm_load_tile_2d(uint32_t lds_addr,
                                                 const void* gptr,
                                                 uint32_t stride_dw)
{
    const uint64_t ga = (uint64_t)(uintptr_t)gptr;
    u32x4 g0;
    g0[0] = 1u;                                                  // count=1
    g0[1] = lds_addr;                                            // LDS byte addr
    g0[2] = (uint32_t)ga;                                        // global[31:0]
    g0[3] = (uint32_t)((ga >> 32) & 0x01FFFFFFu) | (2u << 30);   // [56:32]|type=2
    u32x8 g1;
    g1[0] = (2u << 16)         // data_size = 4B
          | (1u << 20)         // pad_enable
          | (3u << 22)         // pad_interval: 16 DWORDs = 64B
          | (19u << 25);       // pad_amount: 20 DWORDs = 80B
    g1[1] = (16u << 16);       // tensor_dim0 = 16 DWORDs  (lo16)
    g1[2] = (128u << 16);      // tensor_dim0 hi=0 | tensor_dim1 = 128 (lo16)
    g1[3] = (16u << 16);       // tensor_dim1 hi=0 | tile_dim0 = 16 DWORDs
    g1[4] = 128u;              // tile_dim1 = 128 | tile_dim2 = 0
    g1[5] = stride_dw;         // tensor_dim0_stride lo32 (DWORD units)
    g1[6] = 0u;
    g1[7] = 0u;
    asm volatile("tensor_load_to_lds %0, %1" :: "s"(g0), "s"(g1) : "memory");
}

// ---------------------------------------------------------------------------
// GEMM:  C[M,N] = A[M,K] * W[N,K]^T, bf16 operands, fp32 WMMA accumulate.
// Block tile 128x128, 256 threads = 8 wave32, wave tile 32x64 (2x4 subtiles).
// Double-buffered LDS staging, K-loop unrolled by 2 (nk always even here) so
// buffer indices are compile-time constants.
//   USE_TDM = 1: wave 0 issues 2 tensor_load_to_lds per tile (TENSORcnt)
//   USE_TDM = 0: all waves issue global_load_async_to_lds_b128 (ASYNCcnt)
// Requires: M % 128 == 0, K % 64 == 0, W padded to >= ceil(N/128)*128 rows.
// epilogue: 0 = none, 1 = softplus(acc + bias[col])
// ---------------------------------------------------------------------------
template <bool USE_TDM>
__global__ __launch_bounds__(256)
void gemm_bf16_wmma(const __bf16* __restrict__ A, int lda,
                    const __bf16* __restrict__ W, int ldb,
                    float* __restrict__ C, int ldc,
                    int N, int K,
                    int epilogue, const float* __restrict__ bias)
{
    __shared__ __bf16 lA[2][128 * LSTR];
    __shared__ __bf16 lB[2][128 * LSTR];

    const int t      = threadIdx.x;
    const int lane   = t & 31;
    const int wave   = t >> 5;
    const int blockM = blockIdx.y * 128;
    const int blockN = blockIdx.x * 128;
    const int waveM  = (wave >> 1) * 32;
    const int waveN  = (wave & 1) * 64;

    // ---- async staging geometry (128 rows x 64B = 512 x 16B chunks) ----
    uint32_t voffA[2], voffB[2], ldsA[2][2], ldsB[2][2];
    if (!USE_TDM) {
        #pragma unroll
        for (int p = 0; p < 2; ++p) {
            const int c   = t + p * 256;
            const int row = c >> 2;
            const int col = (c & 3) * 16;
            voffA[p] = (uint32_t)(blockM + row) * (uint32_t)(lda * 2) + col;
            voffB[p] = (uint32_t)(blockN + row) * (uint32_t)(ldb * 2) + col;
            #pragma unroll
            for (int b = 0; b < 2; ++b) {
                ldsA[b][p] = (uint32_t)(uintptr_t)(&lA[b][0]) + row * (LSTR * 2) + col;
                ldsB[b][p] = (uint32_t)(uintptr_t)(&lB[b][0]) + row * (LSTR * 2) + col;
            }
        }
    }

    auto issue_tile = [&](int kt, int buf) {
        if (USE_TDM) {
            if (wave == 0) {
                tdm_load_tile_2d((uint32_t)(uintptr_t)(&lA[buf][0]),
                                 A + (size_t)blockM * lda + kt * 32,
                                 (uint32_t)(lda >> 1));
                tdm_load_tile_2d((uint32_t)(uintptr_t)(&lB[buf][0]),
                                 W + (size_t)blockN * ldb + kt * 32,
                                 (uint32_t)(ldb >> 1));
            }
        } else {
            const uint32_t kByte = (uint32_t)kt * 64u;
            #pragma unroll
            for (int p = 0; p < 2; ++p) {
                asm volatile("global_load_async_to_lds_b128 %0, %1, %2"
                             :: "v"(ldsA[buf][p]), "v"(voffA[p] + kByte), "s"(A)
                             : "memory");
                asm volatile("global_load_async_to_lds_b128 %0, %1, %2"
                             :: "v"(ldsB[buf][p]), "v"(voffB[p] + kByte), "s"(W)
                             : "memory");
            }
        }
    };
    auto wait_one = [&]() {   // previous tile landed; next tile still in flight
        if (USE_TDM) __builtin_amdgcn_s_wait_tensorcnt((short)2);
        else         asm volatile("s_wait_asynccnt 4" ::: "memory");
    };
    auto wait_all = [&]() {
        if (USE_TDM) __builtin_amdgcn_s_wait_tensorcnt((short)0);
        else         asm volatile("s_wait_asynccnt 0" ::: "memory");
    };

    v8f acc[2][4];
    {
        v8f z = {};
        #pragma unroll
        for (int m = 0; m < 2; ++m)
            #pragma unroll
            for (int n = 0; n < 4; ++n) acc[m][n] = z;
    }

    // 16-bit A-fragment addressing (ISA 7.12.2):
    //   lanes 0-15: row = lane,    K = {kb..kb+7} U {kb+16..kb+23}, kb = 0
    //   lanes16-31: row = lane-16, same with kb = 8
    const int halfSel  = lane >> 4;
    const int kb       = halfSel * 8;
    const int fragRowA = waveM + (lane & 15);
    const int fragRowB = waveN + (lane & 15);

    auto mma_step = [&](const __bf16* bufA, const __bf16* bufB) {
        v16bf af[2], bf[4];
        #pragma unroll
        for (int m = 0; m < 2; ++m) {
            const __bf16* pa = bufA + (fragRowA + m * 16) * LSTR + kb;
            v8bf lo = *(const v8bf*)pa;
            v8bf hi = *(const v8bf*)(pa + 16);
            #pragma unroll
            for (int i = 0; i < 8; ++i) { af[m][i] = lo[i]; af[m][i + 8] = hi[i]; }
        }
        #pragma unroll
        for (int n = 0; n < 4; ++n) {
            const __bf16* pb = bufB + (fragRowB + n * 16) * LSTR + kb;
            v8bf lo = *(const v8bf*)pb;
            v8bf hi = *(const v8bf*)(pb + 16);
            #pragma unroll
            for (int i = 0; i < 8; ++i) { bf[n][i] = lo[i]; bf[n][i + 8] = hi[i]; }
        }
        #pragma unroll
        for (int m = 0; m < 2; ++m)
            #pragma unroll
            for (int n = 0; n < 4; ++n)
                acc[m][n] = __builtin_amdgcn_wmma_f32_16x16x32_bf16(
                    false, af[m], false, bf[n], (short)0, acc[m][n], false, false);
    };

    const int nk = K >> 5;               // K/32 tiles; even for all call sites
    issue_tile(0, 0);

    for (int kt = 0; kt < nk; kt += 2) {
        // ---- even tile -> buffer 0 (tile kt+1 always exists: nk even) ----
        issue_tile(kt + 1, 1);
        wait_one();
        __syncthreads();
        mma_step(&lA[0][0], &lB[0][0]);
        __syncthreads();

        // ---- odd tile -> buffer 1 ----
        if (kt + 2 < nk) { issue_tile(kt + 2, 0); wait_one(); }
        else             { wait_all(); }
        __syncthreads();
        mma_step(&lA[1][0], &lB[1][0]);
        __syncthreads();
    }

    // ---- store (C/D layout: lane 0-15 -> M+=0, lane 16-31 -> M+=8; VGPR v -> +v) ----
    const int rowBase = blockM + waveM + halfSel * 8;
    const int colBase = blockN + waveN + (lane & 15);
    #pragma unroll
    for (int m = 0; m < 2; ++m) {
        #pragma unroll
        for (int n = 0; n < 4; ++n) {
            const int col = colBase + n * 16;
            if (col < N) {
                const int row0 = rowBase + m * 16;
                #pragma unroll
                for (int v = 0; v < 8; ++v) {
                    float val = acc[m][n][v];
                    if (epilogue == 1) {
                        val += bias[col];
                        val = (val > 20.0f) ? val : logf(1.0f + __expf(val));
                    }
                    C[(size_t)(row0 + v) * ldc + col] = val;
                }
            }
        }
    }
}

// ---------------------------------------------------------------------------
// Depthwise causal conv1d (k=4) + bias + SiLU; fp32 + bf16 outputs.
// ---------------------------------------------------------------------------
__global__ __launch_bounds__(256)
void conv_silu_kernel(const float* __restrict__ xz, const float* __restrict__ cw,
                      const float* __restrict__ cb, float* __restrict__ xs,
                      __bf16* __restrict__ xs_bf)
{
    const int idx = blockIdx.x * 256 + threadIdx.x;      // over B*L*D_INNER
    const int d  = idx & (D_INNER - 1);
    const int bl = idx >> 11;                            // b*L + l
    const int l  = bl & (L_SEQ - 1);

    float acc = cb[d];
    #pragma unroll
    for (int j = 0; j < 4; ++j) {
        const int ls = l - 3 + j;
        if (ls >= 0)
            acc += cw[d * 4 + j] * xz[(size_t)(bl - 3 + j) * (2 * D_INNER) + d];
    }
    const float s = acc / (1.0f + __expf(-acc));         // SiLU
    xs[idx]    = s;
    xs_bf[idx] = (__bf16)s;
}

// ---------------------------------------------------------------------------
// Selective scan: one lane per (b,d) channel, 16-state in registers.
// Fuses y += D*x and SiLU(z) gating; writes over the delta buffer in place.
// ---------------------------------------------------------------------------
__global__ __launch_bounds__(256)
void scan_kernel(const float* __restrict__ xz, const float* __restrict__ xs,
                 const float* __restrict__ dbc, float* __restrict__ dy,
                 const float* __restrict__ A_log, const float* __restrict__ Dp)
{
    const int b = blockIdx.x >> 3;
    const int d = ((blockIdx.x & 7) << 8) + threadIdx.x;

    float An[D_STATE], h[D_STATE];
    #pragma unroll
    for (int n = 0; n < D_STATE; ++n) {
        An[n] = -__expf(A_log[d * D_STATE + n]);
        h[n]  = 0.0f;
    }
    const float Dv = Dp[d];

    __shared__ float sB[D_STATE];
    __shared__ float sC[D_STATE];

    for (int l = 0; l < L_SEQ; ++l) {
        const size_t bl = (size_t)(b * L_SEQ + l);
        if (threadIdx.x < 32) {
            const int n = threadIdx.x;
            const float v = dbc[bl * 96 + DT_RANK + n];
            if (n < D_STATE) sB[n] = v; else sC[n - D_STATE] = v;
        }
        __syncthreads();

        const size_t idx = bl * D_INNER + d;
        const float delta = dy[idx];
        const float xv    = xs[idx];
        const float dx    = delta * xv;

        float y = 0.0f;
        #pragma unroll
        for (int n = 0; n < D_STATE; ++n) {
            const float dA = __expf(delta * An[n]);
            h[n] = dA * h[n] + dx * sB[n];
            y += h[n] * sC[n];
        }
        y += Dv * xv;

        const float zv = xz[bl * (2 * D_INNER) + D_INNER + d];
        dy[idx] = y * (zv / (1.0f + __expf(-zv)));
        __syncthreads();
    }
}

// ---------------------------------------------------------------------------
// Orchestration.
// ---------------------------------------------------------------------------
extern "C" void kernel_launch(void* const* d_in, const int* in_sizes, int n_in,
                              void* d_out, int out_size, void* d_ws, size_t ws_size,
                              hipStream_t stream)
{
    const float* x       = (const float*)d_in[0];
    const float* in_proj = (const float*)d_in[1];
    const float* conv_w  = (const float*)d_in[2];
    const float* conv_b  = (const float*)d_in[3];
    const float* x_proj  = (const float*)d_in[4];
    const float* dt_w    = (const float*)d_in[5];
    const float* dt_b    = (const float*)d_in[6];
    const float* A_log   = (const float*)d_in[7];
    const float* Dp      = (const float*)d_in[8];
    const float* out_w   = (const float*)d_in[9];
    float* out = (float*)d_out;

    char* ws = (char*)d_ws;
    float*  xz     = (float*)(ws);                                // 2048*4096 f32
    float*  xs     = (float*)(ws + (size_t)33554432);             // 2048*2048 f32
    float*  dbc    = (float*)(ws + (size_t)50331648);             // 2048*96   f32
    float*  dy     = (float*)(ws + (size_t)51118080);             // 2048*2048 f32
    __bf16* xbf    = (__bf16*)(ws + (size_t)67895296);            // 2048*1024
    __bf16* w_in   = (__bf16*)(ws + (size_t)72089600);            // 4096*1024
    __bf16* xs_bf  = (__bf16*)(ws + (size_t)80478208);            // 2048*2048
    __bf16* w_xp   = (__bf16*)(ws + (size_t)88866816);            // 128*2048 (pad)
    __bf16* dbc_bf = (__bf16*)(ws + (size_t)89391104);            // 2048*96
    __bf16* w_dt   = (__bf16*)(ws + (size_t)89784320);            // 2048*64
    __bf16* ybf    = (__bf16*)(ws + (size_t)90046464);            // 2048*2048
    __bf16* w_out  = (__bf16*)(ws + (size_t)98435072);            // 1024*2048

    #define PACK(src, dst, cnt, pad) \
        pack_bf16<<<((pad) + 255) / 256, 256, 0, stream>>>((src), (dst), (cnt), (pad))

    PACK(x,       xbf,  M_ROWS * D_MODEL,      M_ROWS * D_MODEL);
    PACK(in_proj, w_in, 2 * D_INNER * D_MODEL, 2 * D_INNER * D_MODEL);

    // 1) in_proj GEMM (TDM staging): (2048,1024) x (4096,1024)^T -> xz
    gemm_bf16_wmma<true><<<dim3(2 * D_INNER / 128, M_ROWS / 128), 256, 0, stream>>>(
        xbf, D_MODEL, w_in, D_MODEL, xz, 2 * D_INNER, 2 * D_INNER, D_MODEL, 0, nullptr);

    // 2) depthwise conv + SiLU
    conv_silu_kernel<<<(M_ROWS * D_INNER) / 256, 256, 0, stream>>>(
        xz, conv_w, conv_b, xs, xs_bf);

    // 3) x_proj GEMM (async staging): (2048,2048) x (96,2048)^T -> dbc
    PACK(x_proj, w_xp, 96 * D_INNER, 128 * D_INNER);
    gemm_bf16_wmma<false><<<dim3(1, M_ROWS / 128), 256, 0, stream>>>(
        xs_bf, D_INNER, w_xp, D_INNER, dbc, 96, 96, D_INNER, 0, nullptr);

    // 4) dt_proj + softplus (async staging): (2048,64) x (2048,64)^T -> dy
    PACK(dbc,  dbc_bf, M_ROWS * 96,       M_ROWS * 96);
    PACK(dt_w, w_dt,   D_INNER * DT_RANK, D_INNER * DT_RANK);
    gemm_bf16_wmma<false><<<dim3(D_INNER / 128, M_ROWS / 128), 256, 0, stream>>>(
        dbc_bf, 96, w_dt, DT_RANK, dy, D_INNER, D_INNER, DT_RANK, 1, dt_b);

    // 5) selective scan + skip + gate (in place over dy)
    scan_kernel<<<B_SZ * (D_INNER / 256), 256, 0, stream>>>(xz, xs, dbc, dy, A_log, Dp);

    // 6) out_proj GEMM (TDM staging): (2048,2048) x (1024,2048)^T -> out
    PACK(dy,    ybf,   M_ROWS * D_INNER,  M_ROWS * D_INNER);
    PACK(out_w, w_out, D_MODEL * D_INNER, D_MODEL * D_INNER);
    gemm_bf16_wmma<true><<<dim3(D_MODEL / 128, M_ROWS / 128), 256, 0, stream>>>(
        ybf, D_INNER, w_out, D_INNER, out, D_MODEL, D_MODEL, D_INNER, 0, nullptr);

    #undef PACK
}
